// MolGraphPrior_2422361554990
// MI455X (gfx1250) — matrix operand order repaired and checked by reading, not compiled, not run on hardware
//
#include <hip/hip_runtime.h>
#include <hip/hip_bf16.h>

// ---------------------------------------------------------------------------
// MolGraphPrior on MI455X (gfx1250, wave32, WMMA).
//
// d_in flattening assumption (setup_inputs() dict insertion order, recursive):
//   0 prot_x[32768,11] 1 prot_edge_index[2,262144] 2 prot_edge_attr[262144,13]
//   3 prot_batch 4 drug_x[4096,20] 5 drug_edge_index[2,16384]
//   6 drug_edge_attr[16384,5] 7 drug_batch
//   prot_params @8 : node_W,node_b,edge_W,edge_b,
//     convs l=0..3: (12+4l)W1[512,256] (13+4l)b1 (14+4l)W2[256,512] (15+4l)b2
//     bns   l=0..3: (28+4l)g (29+4l)b (30+4l)rm (31+4l)rv
//     44 gate_W 45 gate_b 46 pool_W 47 pool_b
//   drug_params @48 (same layout)
//   88 in_W[768,256] 89 in_b[768] 90 out_W[256,256] 91 out_b[256]
//   92..95 pool_ctx (gate_W,gate_b,pool_W,pool_b)
//   96 W1[256,512] 97 b1 98 W2[1,256] 99 b2
// ---------------------------------------------------------------------------

#define HID 256

typedef __attribute__((ext_vector_type(16))) __bf16 v16bf;
typedef __attribute__((ext_vector_type(8)))  __bf16 v8bf;
typedef __attribute__((ext_vector_type(4)))  __bf16 v4bf;
typedef __attribute__((ext_vector_type(8)))  float  v8f;

__device__ __forceinline__ float nan2num(float v) {
  v = (v != v) ? 0.0f : v;
  v = fminf(v,  3.4028234663852886e38f);
  v = fmaxf(v, -3.4028234663852886e38f);
  return v;
}

// ---------------- node / edge feature encoder (tiny-K, scalar) -------------
template<int FIN, int BF16OUT>
__global__ void feat_enc_kernel(const float* __restrict__ xin,
                                const float* __restrict__ W,
                                const float* __restrict__ b,
                                void* __restrict__ out) {
  __shared__ float xr[FIN];
  const int n = blockIdx.x;
  const int c = threadIdx.x;
  if (c < FIN) xr[c] = nan2num(xin[(long)n * FIN + c]);
  __syncthreads();
  float acc = b[c];
  const float* w = W + c * FIN;
#pragma unroll
  for (int k = 0; k < FIN; ++k) acc += xr[k] * w[k];
  acc = fmaxf(acc, 0.0f);
  if (BF16OUT) ((__bf16*)out)[(long)n * HID + c] = (__bf16)acc;
  else         ((float*) out)[(long)n * HID + c] = acc;
}

// ---------------- zero fill -------------------------------------------------
__global__ void zero_kernel(float4* __restrict__ p, long n4) {
  long i = blockIdx.x * (long)blockDim.x + threadIdx.x;
  if (i < n4) p[i] = make_float4(0.f, 0.f, 0.f, 0.f);
}

// ---------------- f32 -> bf16 weight conversion -----------------------------
__global__ void cvt_bf16_kernel(const float* __restrict__ s, __bf16* __restrict__ d, long n) {
  long i = blockIdx.x * (long)blockDim.x + threadIdx.x;
  if (i < n) d[i] = (__bf16)s[i];
}

// ---------------- GINE message + scatter-add aggregation --------------------
// msg = relu(x[src] + e); aggr[dst] += msg. 64 threads/edge, 4 feats/thread.
__global__ void msg_aggr_kernel(const float* __restrict__ x, const __bf16* __restrict__ e,
                                const int* __restrict__ src, const int* __restrict__ dst,
                                float* __restrict__ aggr, int E) {
  long tid  = blockIdx.x * (long)blockDim.x + threadIdx.x;
  int edge  = (int)(tid >> 6);
  if (edge >= E) return;
  int f = ((int)tid & 63) << 2;
  int s = src[edge], d = dst[edge];
  float4 xv = *(const float4*)(x + (long)s * HID + f);
  v4bf  ev = *(const v4bf*)(e + (long)edge * HID + f);
  float* a = aggr + (long)d * HID + f;
  unsafeAtomicAdd(a + 0, fmaxf(xv.x + (float)ev[0], 0.f));
  unsafeAtomicAdd(a + 1, fmaxf(xv.y + (float)ev[1], 0.f));
  unsafeAtomicAdd(a + 2, fmaxf(xv.z + (float)ev[2], 0.f));
  unsafeAtomicAdd(a + 3, fmaxf(xv.w + (float)ev[3], 0.f));
}

// ---------------- generic WMMA GEMM ----------------------------------------
// Out = EPI( A[M,K] @ Bw[N,K]^T + bias )  (Bw row-major [N,K] bf16;
// BTRANS=1 means Bw stored K-major: B[k,n] = Bw[k*ldb + n]).
// AMODE: 0 = A f32, 1 = A bf16, 2 = A f32 + A2 f32 (elementwise sum, GINE h)
// EPI:   0 = (+bias)*scale -> f32      (scores / out-proj)
//        1 = relu(+bias)   -> bf16     (GIN MLP hidden)
//        2 = BN(+bias)+relu-> f32      (GIN MLP out)
//        3 = (+bias)       -> bf16     (qkv / ctx)
// One wave computes a 16x64 tile (4 accumulators). 4 waves/block along M.
// z-batching: z -> (b = z/heads, h = z%heads) with separate strides.
// A data streams (L2/HBM) -> prefetch next k-step; B (weights) is L2-hot.
template<int AMODE, int BTRANS, int EPI>
__global__ void __launch_bounds__(128) wmma_gemm(
    const void* __restrict__ Ap, const float* __restrict__ A2, long lda,
    const __bf16* __restrict__ Bw, long ldb,
    void* __restrict__ Outp, long ldo,
    const float* __restrict__ bias,
    const float* __restrict__ bn_g, const float* __restrict__ bn_b,
    const float* __restrict__ bn_rm, const float* __restrict__ bn_rv,
    int K, int heads,
    long A_b, long A_h, long B_b, long B_h, long O_b, long O_h,
    float scale) {
  const int lane = threadIdx.x & 31;
  const int wave = threadIdx.x >> 5;
  const int n0 = blockIdx.x * 64;
  const int m0 = (blockIdx.y * 4 + wave) * 16;
  const int z  = blockIdx.z;
  const int bz = z / heads, hz = z - bz * heads;
  const long aoff = (long)bz * A_b + (long)hz * A_h;
  const long boff = (long)bz * B_b + (long)hz * B_h;
  const long ooff = (long)bz * O_b + (long)hz * O_h;

  const int r = lane & 15;        // row (A) / col (B,C) within 16
  const int h = lane >> 4;        // half-wave selector

  v8f acc[4];
#pragma unroll
  for (int t = 0; t < 4; ++t)
#pragma unroll
    for (int j = 0; j < 8; ++j) acc[t][j] = 0.0f;

  const long arow = aoff + (long)(m0 + r) * lda + h * 8;

  for (int kb = 0; kb < K; kb += 32) {
    // software prefetch of the next k-step's A runs (global_prefetch_b8)
    if (kb + 32 < K) {
      if (AMODE == 1) {
        __builtin_prefetch((const __bf16*)Ap + arow + kb + 32, 0, 1);
      } else {
        __builtin_prefetch((const float*)Ap + arow + kb + 32, 0, 1);
        if (AMODE == 2) __builtin_prefetch(A2 + arow + kb + 32, 0, 1);
      }
    }

    // ---- A fragment: 16x32 bf16, per ISA layout. Per lane two contiguous
    // 8-element runs at K offsets h*8 and 16+h*8.
    v16bf a;
    if (AMODE == 1) {
      const __bf16* Ab = (const __bf16*)Ap + arow + kb;
      v8bf a0 = *(const v8bf*)(Ab);
      v8bf a1 = *(const v8bf*)(Ab + 16);
#pragma unroll
      for (int q = 0; q < 8; ++q) { a[q] = a0[q]; a[8 + q] = a1[q]; }
    } else {
      const float* Af = (const float*)Ap + arow + kb;
      float4 f0 = *(const float4*)(Af + 0);
      float4 f1 = *(const float4*)(Af + 4);
      float4 f2 = *(const float4*)(Af + 16);
      float4 f3 = *(const float4*)(Af + 20);
      if (AMODE == 2) {
        const float* Ag = A2 + arow + kb;
        float4 g0 = *(const float4*)(Ag + 0);
        float4 g1 = *(const float4*)(Ag + 4);
        float4 g2 = *(const float4*)(Ag + 16);
        float4 g3 = *(const float4*)(Ag + 20);
        f0.x += g0.x; f0.y += g0.y; f0.z += g0.z; f0.w += g0.w;
        f1.x += g1.x; f1.y += g1.y; f1.z += g1.z; f1.w += g1.w;
        f2.x += g2.x; f2.y += g2.y; f2.z += g2.z; f2.w += g2.w;
        f3.x += g3.x; f3.y += g3.y; f3.z += g3.z; f3.w += g3.w;
      }
      a[0]  = (__bf16)f0.x; a[1]  = (__bf16)f0.y; a[2]  = (__bf16)f0.z; a[3]  = (__bf16)f0.w;
      a[4]  = (__bf16)f1.x; a[5]  = (__bf16)f1.y; a[6]  = (__bf16)f1.z; a[7]  = (__bf16)f1.w;
      a[8]  = (__bf16)f2.x; a[9]  = (__bf16)f2.y; a[10] = (__bf16)f2.z; a[11] = (__bf16)f2.w;
      a[12] = (__bf16)f3.x; a[13] = (__bf16)f3.y; a[14] = (__bf16)f3.z; a[15] = (__bf16)f3.w;
    }

    // ---- B fragments: 32x16 bf16, column n = lane&15, same K order as A:
    // two 8-element runs at K offsets h*8 and 16+h*8.
    v16bf bt[4];
#pragma unroll
    for (int t = 0; t < 4; ++t) {
      if (!BTRANS) {
        const __bf16* Wp = Bw + boff + (long)(n0 + t * 16 + r) * ldb + kb + h * 8;
        v8bf b0 = *(const v8bf*)(Wp);
        v8bf b1 = *(const v8bf*)(Wp + 16);
#pragma unroll
        for (int q = 0; q < 8; ++q) { bt[t][q] = b0[q]; bt[t][8 + q] = b1[q]; }
      } else {
        const __bf16* Wp = Bw + boff + (long)(n0 + t * 16 + r);
#pragma unroll
        for (int v = 0; v < 8; ++v) {
          long kk = kb + h * 8 + 2 * v + (v >= 4 ? 8 : 0);
          bt[t][2 * v]     = Wp[kk * ldb];
          bt[t][2 * v + 1] = Wp[(kk + 1) * ldb];
        }
      }
    }

#pragma unroll
    for (int t = 0; t < 4; ++t)
      acc[t] = __builtin_amdgcn_wmma_f32_16x16x32_bf16(
          false, a, false, bt[t], (short)0, acc[t], false, false);
  }

  // ---- epilogue. C/D layout: VGPR j -> row m0+j+8h, col n0+t*16+(lane&15).
#pragma unroll
  for (int t = 0; t < 4; ++t) {
#pragma unroll
    for (int j = 0; j < 8; ++j) {
      const int row = m0 + j + 8 * h;
      const int col = n0 + t * 16 + r;
      float val = acc[t][j];
      if (EPI == 0) {
        if (bias) val += bias[col];
        val *= scale;
        ((float*)Outp)[ooff + (long)row * ldo + col] = val;
      } else if (EPI == 1) {
        val = fmaxf(val + bias[col], 0.0f);
        ((__bf16*)Outp)[ooff + (long)row * ldo + col] = (__bf16)val;
      } else if (EPI == 2) {
        val += bias[col];
        val = (val - bn_rm[col]) * rsqrtf(bn_rv[col] + 1e-5f) * bn_g[col] + bn_b[col];
        val = fmaxf(val, 0.0f);
        ((float*)Outp)[ooff + (long)row * ldo + col] = val;
      } else { // EPI == 3
        if (bias) val += bias[col];
        ((__bf16*)Outp)[ooff + (long)row * ldo + col] = (__bf16)val;
      }
    }
  }
}

// ---------------- attention-pool: per-node gate = x . gW + gb ---------------
__global__ void gate_kernel(const float* __restrict__ x, const float* __restrict__ gW,
                            const float* __restrict__ gb, float* __restrict__ gate, int N) {
  int node = blockIdx.x * (blockDim.x >> 5) + (threadIdx.x >> 5);
  int lane = threadIdx.x & 31;
  if (node >= N) return;
  const float* xr = x + (long)node * HID;
  float acc = 0.0f;
#pragma unroll
  for (int k = lane; k < HID; k += 32) acc += xr[k] * gW[k];
#pragma unroll
  for (int o = 16; o; o >>= 1) acc += __shfl_xor(acc, o, 32);
  if (lane == 0) gate[node] = acc + gb[0];
}

// softmax(gate) over a contiguous graph block, s[b] = sum_i alpha_i * x_i
__global__ void att_pool_ws_kernel(const float* __restrict__ x, const float* __restrict__ gate,
                                   float* __restrict__ s, int P) {
  __shared__ float red[256];
  const int b = blockIdx.x, c = threadIdx.x;
  const float* gb_ = gate + (long)b * P;
  float lm = -3.4e38f;
  for (int i = c; i < P; i += 256) lm = fmaxf(lm, gb_[i]);
  red[c] = lm; __syncthreads();
  for (int o = 128; o; o >>= 1) { if (c < o) red[c] = fmaxf(red[c], red[c + o]); __syncthreads(); }
  const float m = red[0]; __syncthreads();
  float ls = 0.0f;
  for (int i = c; i < P; i += 256) ls += expf(gb_[i] - m);
  red[c] = ls; __syncthreads();
  for (int o = 128; o; o >>= 1) { if (c < o) red[c] += red[c + o]; __syncthreads(); }
  const float zinv = 1.0f / red[0];
  float acc = 0.0f;
  const float* xb = x + (long)b * P * HID;
  for (int i = 0; i < P; ++i) acc += expf(gb_[i] - m) * xb[(long)i * HID + c];
  s[b * HID + c] = acc * zinv;
}

// pooled[b][c] = s[b] . pW[c] + pb[c]   (64 x 256, K=256 -- tiny)
__global__ void small_gemm_kernel(const float* __restrict__ s, const float* __restrict__ pW,
                                  const float* __restrict__ pb, float* __restrict__ out) {
  const int b = blockIdx.x, c = threadIdx.x;
  float acc = pb[c];
  const float* sb = s + b * HID;
  const float* w = pW + c * HID;
#pragma unroll 8
  for (int k = 0; k < HID; ++k) acc += sb[k] * w[k];
  out[b * HID + c] = acc;
}

// ---------------- row softmax over 512 (attention scores, in place) --------
__global__ void softmax512_kernel(float* __restrict__ S) {
  __shared__ float red[128];
  float* row = S + (long)blockIdx.x * 512;
  const int t = threadIdx.x;
  float lm = -3.4e38f;
  for (int i = t; i < 512; i += 128) lm = fmaxf(lm, row[i]);
  red[t] = lm; __syncthreads();
  for (int o = 64; o; o >>= 1) { if (t < o) red[t] = fmaxf(red[t], red[t + o]); __syncthreads(); }
  const float m = red[0]; __syncthreads();
  float ls = 0.0f;
  for (int i = t; i < 512; i += 128) { float e = expf(row[i] - m); row[i] = e; ls += e; }
  red[t] = ls; __syncthreads();
  for (int o = 64; o; o >>= 1) { if (t < o) red[t] += red[t + o]; __syncthreads(); }
  const float inv = 1.0f / red[0];
  for (int i = t; i < 512; i += 128) row[i] *= inv;
}

// ---------------- final scorer: relu(concat@W1^T+b1)@W2^T+b2 ----------------
__global__ void scorer_kernel(const float* __restrict__ pemb, const float* __restrict__ demb,
                              const float* __restrict__ W1, const float* __restrict__ b1,
                              const float* __restrict__ W2, const float* __restrict__ b2,
                              float* __restrict__ out) {
  __shared__ float red[256];
  const int b = blockIdx.x, c = threadIdx.x;
  float acc = b1[c];
  const float* w  = W1 + (long)c * 512;
  const float* pe = pemb + b * HID;
  const float* de = demb + b * HID;
#pragma unroll 8
  for (int k = 0; k < HID; ++k) acc += pe[k] * w[k];
#pragma unroll 8
  for (int k = 0; k < HID; ++k) acc += de[k] * w[HID + k];
  red[c] = fmaxf(acc, 0.0f) * W2[c];
  __syncthreads();
  for (int o = 128; o; o >>= 1) { if (c < o) red[c] += red[c + o]; __syncthreads(); }
  if (c == 0) out[b] = red[0] + b2[0];
}

// ===========================================================================
extern "C" void kernel_launch(void* const* d_in, const int* in_sizes, int n_in,
                              void* d_out, int out_size, void* d_ws, size_t ws_size,
                              hipStream_t stream) {
  (void)in_sizes; (void)n_in; (void)out_size; (void)ws_size;
  const int Np = 32768, Ep = 262144, Nd = 4096, Ed = 16384;

  const float* prot_x_in = (const float*)d_in[0];
  const int*   p_ei      = (const int*)  d_in[1];
  const float* p_ea      = (const float*)d_in[2];
  const float* drug_x_in = (const float*)d_in[4];
  const int*   d_ei      = (const int*)  d_in[5];
  const float* d_ea      = (const float*)d_in[6];
  auto F = [&](int i) { return (const float*)d_in[i]; };

  // ---- workspace carve-out (aligned 256B); ~265 MB total ----
  char* wsp = (char*)d_ws;
  size_t off = 0;
  auto alloc = [&](size_t bytes) -> void* {
    void* p = wsp + off;
    off += (bytes + 255) & ~(size_t)255;
    return p;
  };
  __bf16* pe    = (__bf16*)alloc((size_t)Ep * HID * 2);   // 128 MB (fits L2)
  float*  px    = (float*) alloc((size_t)Np * HID * 4);   // 32 MB
  float*  paggr = (float*) alloc((size_t)Np * HID * 4);   // 32 MB (later: scores)
  __bf16* ph2   = (__bf16*)alloc((size_t)Np * 512 * 2);   // 32 MB (later: k|v)
  __bf16* de    = (__bf16*)alloc((size_t)Ed * HID * 2);
  float*  dx    = (float*) alloc((size_t)Nd * HID * 4);
  float*  daggr = (float*) alloc((size_t)Nd * HID * 4);   // later: ctx bf16
  __bf16* dh2   = (__bf16*)alloc((size_t)Nd * 512 * 2);   // later: q bf16
  float*  dctx  = (float*) alloc((size_t)Nd * HID * 4);
  float*  gates = (float*) alloc((size_t)Np * 4);
  float*  sbuf  = (float*) alloc(64 * HID * 4);
  float*  pemb  = (float*) alloc(64 * HID * 4);
  float*  demb  = (float*) alloc(64 * HID * 4);
  __bf16 *pW1bf[4], *pW2bf[4], *dW1bf[4], *dW2bf[4];
  for (int l = 0; l < 4; ++l) {
    pW1bf[l] = (__bf16*)alloc(512 * 256 * 2);
    pW2bf[l] = (__bf16*)alloc(256 * 512 * 2);
    dW1bf[l] = (__bf16*)alloc(512 * 256 * 2);
    dW2bf[l] = (__bf16*)alloc(256 * 512 * 2);
  }
  __bf16* inWbf  = (__bf16*)alloc(768 * 256 * 2);
  __bf16* outWbf = (__bf16*)alloc(256 * 256 * 2);
  // aliases (lifetimes disjoint in stream order)
  float*  scores = paggr;                 // [256 batches][64][512] f32 = 32 MB
  __bf16* kbuf   = ph2;                   // [32768,256] bf16
  __bf16* vbuf   = ph2 + (size_t)Np * HID;
  __bf16* qbuf   = dh2;                   // [4096,256] bf16
  __bf16* ctx    = (__bf16*)daggr;        // [4096,256] bf16

  auto cvt = [&](const float* s, __bf16* dd, long n) {
    cvt_bf16_kernel<<<(unsigned)((n + 255) / 256), 256, 0, stream>>>(s, dd, n);
  };
  for (int l = 0; l < 4; ++l) {
    cvt(F(12 + 4 * l), pW1bf[l], 512 * 256);
    cvt(F(14 + 4 * l), pW2bf[l], 256 * 512);
    cvt(F(52 + 4 * l), dW1bf[l], 512 * 256);
    cvt(F(54 + 4 * l), dW2bf[l], 256 * 512);
  }
  cvt(F(88), inWbf, 768 * 256);
  cvt(F(90), outWbf, 256 * 256);

  // ================= protein encoder =================
  feat_enc_kernel<11, 0><<<Np, 256, 0, stream>>>(prot_x_in, F(8), F(9), px);
  feat_enc_kernel<13, 1><<<Ep, 256, 0, stream>>>(p_ea, F(10), F(11), pe);
  for (int l = 0; l < 4; ++l) {
    zero_kernel<<<(Np * HID / 4 + 255) / 256, 256, 0, stream>>>((float4*)paggr, (long)Np * HID / 4);
    msg_aggr_kernel<<<Ep * 64 / 256, 256, 0, stream>>>(px, pe, p_ei, p_ei + Ep, paggr, Ep);
    wmma_gemm<2, 0, 1><<<dim3(8, Np / 64, 1), 128, 0, stream>>>(
        px, paggr, 256, pW1bf[l], 256, ph2, 512, F(13 + 4 * l),
        nullptr, nullptr, nullptr, nullptr, 256, 1, 0, 0, 0, 0, 0, 0, 1.0f);
    wmma_gemm<1, 0, 2><<<dim3(4, Np / 64, 1), 128, 0, stream>>>(
        ph2, nullptr, 512, pW2bf[l], 512, px, 256, F(15 + 4 * l),
        F(28 + 4 * l), F(29 + 4 * l), F(30 + 4 * l), F(31 + 4 * l),
        512, 1, 0, 0, 0, 0, 0, 0, 1.0f);
  }
  // protein attentional pooling -> p_emb
  gate_kernel<<<Np / 8, 256, 0, stream>>>(px, F(44), F(45), gates, Np);
  att_pool_ws_kernel<<<64, 256, 0, stream>>>(px, gates, sbuf, 512);
  small_gemm_kernel<<<64, 256, 0, stream>>>(sbuf, F(46), F(47), pemb);

  // ================= drug encoder (pooled output unused by reference) =====
  feat_enc_kernel<20, 0><<<Nd, 256, 0, stream>>>(drug_x_in, F(48), F(49), dx);
  feat_enc_kernel<5, 1><<<Ed, 256, 0, stream>>>(d_ea, F(50), F(51), de);
  for (int l = 0; l < 4; ++l) {
    zero_kernel<<<(Nd * HID / 4 + 255) / 256, 256, 0, stream>>>((float4*)daggr, (long)Nd * HID / 4);
    msg_aggr_kernel<<<Ed * 64 / 256, 256, 0, stream>>>(dx, de, d_ei, d_ei + Ed, daggr, Ed);
    wmma_gemm<2, 0, 1><<<dim3(8, Nd / 64, 1), 128, 0, stream>>>(
        dx, daggr, 256, dW1bf[l], 256, dh2, 512, F(53 + 4 * l),
        nullptr, nullptr, nullptr, nullptr, 256, 1, 0, 0, 0, 0, 0, 0, 1.0f);
    wmma_gemm<1, 0, 2><<<dim3(4, Nd / 64, 1), 128, 0, stream>>>(
        dh2, nullptr, 512, dW2bf[l], 512, dx, 256, F(55 + 4 * l),
        F(68 + 4 * l), F(69 + 4 * l), F(70 + 4 * l), F(71 + 4 * l),
        512, 1, 0, 0, 0, 0, 0, 0, 1.0f);
  }

  // ================= cross attention (drug queries protein) ===============
  const float* in_b = F(89);
  wmma_gemm<0, 0, 3><<<dim3(4, Nd / 64, 1), 128, 0, stream>>>(        // Q
      dx, nullptr, 256, inWbf, 256, qbuf, 256, in_b,
      nullptr, nullptr, nullptr, nullptr, 256, 1, 0, 0, 0, 0, 0, 0, 1.0f);
  wmma_gemm<0, 0, 3><<<dim3(4, Np / 64, 1), 128, 0, stream>>>(        // K
      px, nullptr, 256, inWbf + 256 * 256, 256, kbuf, 256, in_b + 256,
      nullptr, nullptr, nullptr, nullptr, 256, 1, 0, 0, 0, 0, 0, 0, 1.0f);
  wmma_gemm<0, 0, 3><<<dim3(4, Np / 64, 1), 128, 0, stream>>>(        // V
      px, nullptr, 256, inWbf + 512 * 256, 256, vbuf, 256, in_b + 512,
      nullptr, nullptr, nullptr, nullptr, 256, 1, 0, 0, 0, 0, 0, 0, 1.0f);
  // scores[b,h] = Q_bh[64,64] @ K_bh[512,64]^T * 1/8 ; z = b*4 + h
  wmma_gemm<1, 0, 0><<<dim3(8, 1, 256), 128, 0, stream>>>(
      qbuf, nullptr, 256, kbuf, 256, scores, 512, nullptr,
      nullptr, nullptr, nullptr, nullptr, 64, 4,
      64L * 256, 64, 512L * 256, 64, 4L * 64 * 512, 64L * 512, 0.125f);
  softmax512_kernel<<<256 * 64, 128, 0, stream>>>(scores);
  // ctx[b,h] = P[64,512] @ V_bh[512,64]   (B stored K-major -> BTRANS)
  wmma_gemm<0, 1, 3><<<dim3(1, 1, 256), 128, 0, stream>>>(
      scores, nullptr, 512, vbuf, 256, ctx, 256, nullptr,
      nullptr, nullptr, nullptr, nullptr, 512, 4,
      4L * 64 * 512, 64L * 512, 512L * 256, 64, 64L * 256, 64, 1.0f);
  // out projection -> d_ctx f32 [4096,256]
  wmma_gemm<1, 0, 0><<<dim3(4, Nd / 64, 1), 128, 0, stream>>>(
      ctx, nullptr, 256, outWbf, 256, dctx, 256, F(91),
      nullptr, nullptr, nullptr, nullptr, 256, 1, 0, 0, 0, 0, 0, 0, 1.0f);

  // ================= context pooling -> d_emb ==============================
  gate_kernel<<<Nd / 8, 256, 0, stream>>>(dctx, F(92), F(93), gates, Nd);
  att_pool_ws_kernel<<<64, 256, 0, stream>>>(dctx, gates, sbuf, 64);
  small_gemm_kernel<<<64, 256, 0, stream>>>(sbuf, F(94), F(95), demb);

  // ================= scorer ================================================
  scorer_kernel<<<64, 256, 0, stream>>>(pemb, demb, F(96), F(97), F(98), F(99),
                                        (float*)d_out);
}
